// PrototypeLayer_2826088481484
// MI455X (gfx1250) — matrix operand order repaired and checked by reading, not compiled, not run on hardware
//
#include <hip/hip_runtime.h>
#include <math.h>

typedef __attribute__((ext_vector_type(16))) _Float16 v16h;
typedef __attribute__((ext_vector_type(8)))  _Float16 v8h;
typedef __attribute__((ext_vector_type(8)))  float    v8f;

#define WMMA_F16(A, B, C) \
  __builtin_amdgcn_wmma_f32_16x16x32_f16(false, (A), false, (B), (short)0, (C), false, false)

// ---------------------------------------------------------------------------
// Prep: v = src[row,k]*w[k]; split into f16 hi/lo; exact f32 row norm Σv².
// One block per row, blockDim.x == D (256).
// ---------------------------------------------------------------------------
__global__ void prep_split_kernel(const float* __restrict__ src,
                                  const float* __restrict__ w,
                                  _Float16* __restrict__ hi,
                                  _Float16* __restrict__ lo,
                                  float* __restrict__ norm,
                                  int D) {
  __shared__ float sdata[256];
  const int row = blockIdx.x;
  const int tid = threadIdx.x;
  const size_t idx = (size_t)row * D + tid;
  const float v = src[idx] * w[tid];
  const _Float16 h = (_Float16)v;
  hi[idx] = h;
  lo[idx] = (_Float16)(v - (float)h);
  sdata[tid] = v * v;
  __syncthreads();
  for (int s = 128; s > 0; s >>= 1) {
    if (tid < s && tid + s < D) sdata[tid] += sdata[tid + s];
    __syncthreads();
  }
  if (tid == 0) norm[row] = sdata[0];
}

// ---------------------------------------------------------------------------
// Fragment loaders (row-major [rows, D] f16 arrays).
// A-matrix 16x32 f16 layout (ISA 7.12.2): lanes 0-15 hold row=lane,
//   K = {0..7} (VGPR0-3) and {16..23} (VGPR4-7); lanes 16-31: K = {8..15},{24..31}.
// B-matrix 32x16: lane = column; same K order as A (lanes 0-15 K={0..7},{16..23}; lanes 16-31 K={8..15},{24..31}).
// ---------------------------------------------------------------------------
__device__ __forceinline__ v16h load_a_frag(const _Float16* __restrict__ base,
                                            int D, int row, int k0, int half) {
  const _Float16* p = base + (size_t)row * D + k0 + half * 8;
  v8h p0 = *(const v8h*)(p);        // K = k0 + half*8 + {0..7}
  v8h p1 = *(const v8h*)(p + 16);   // K = k0 + 16 + half*8 + {0..7}
  return __builtin_shufflevector(p0, p1, 0,1,2,3,4,5,6,7,8,9,10,11,12,13,14,15);
}

__device__ __forceinline__ v16h load_b_frag(const _Float16* __restrict__ base,
                                            int D, int row, int k0, int half) {
  const _Float16* p = base + (size_t)row * D + k0 + half * 8;
  v8h p0 = *(const v8h*)(p);        // K = k0 + half*8 + {0..7}
  v8h p1 = *(const v8h*)(p + 16);   // K = k0 + 16 + half*8 + {0..7}
  return __builtin_shufflevector(p0, p1, 0,1,2,3,4,5,6,7,8,9,10,11,12,13,14,15);
}

// ---------------------------------------------------------------------------
// GEMM + epilogue. 8 waves/block; wave tile = 32x32 (2x2 of 16x16 WMMA);
// workgroup tile = 128 rows x 64 cols. hi/lo f16 split: 3 WMMAs per tile pair.
// ---------------------------------------------------------------------------
__global__ void __launch_bounds__(256)
proto_wmma_kernel(const _Float16* __restrict__ xhi, const _Float16* __restrict__ xlo,
                  const _Float16* __restrict__ phi, const _Float16* __restrict__ plo,
                  const float* __restrict__ xnorm, const float* __restrict__ pnorm,
                  const float* __restrict__ tptr,
                  float* __restrict__ out, int N, int M, int D) {
  const int lane = threadIdx.x & 31;
  const int wave = threadIdx.x >> 5;
  const int half = lane >> 4;
  const int l15  = lane & 15;
  const int row_base = blockIdx.x * 128 + (wave >> 1) * 32;
  const int col_base = blockIdx.y * 64  + (wave & 1) * 32;

  v8f acc[2][2];
#pragma unroll
  for (int i = 0; i < 2; ++i)
#pragma unroll
    for (int j = 0; j < 2; ++j)
#pragma unroll
      for (int t = 0; t < 8; ++t) acc[i][j][t] = 0.0f;

  for (int k0 = 0; k0 < D; k0 += 32) {
    // streaming-x prefetch for the K-tile after next (prototypes stay in L2);
    // locality 3 -> near-cache (WGP) temporal policy.
    if (k0 + 64 < D) {
      __builtin_prefetch(xhi + (size_t)(row_base + l15) * D + k0 + 64, 0, 3);
      __builtin_prefetch(xhi + (size_t)(row_base + 16 + l15) * D + k0 + 64, 0, 3);
    }

    v16h ahi[2], alo[2], bhi[2], blo[2];
#pragma unroll
    for (int i = 0; i < 2; ++i) {
      ahi[i] = load_a_frag(xhi, D, row_base + i * 16 + l15, k0, half);
      alo[i] = load_a_frag(xlo, D, row_base + i * 16 + l15, k0, half);
    }
#pragma unroll
    for (int j = 0; j < 2; ++j) {
      bhi[j] = load_b_frag(phi, D, col_base + j * 16 + l15, k0, half);
      blo[j] = load_b_frag(plo, D, col_base + j * 16 + l15, k0, half);
    }

#pragma unroll
    for (int i = 0; i < 2; ++i)
#pragma unroll
      for (int j = 0; j < 2; ++j) {
        acc[i][j] = WMMA_F16(ahi[i], bhi[j], acc[i][j]);  // hi*hi
        acc[i][j] = WMMA_F16(ahi[i], blo[j], acc[i][j]);  // hi*lo
        acc[i][j] = WMMA_F16(alo[i], bhi[j], acc[i][j]);  // lo*hi
      }
  }

  // Epilogue: D-matrix element (VGPR t, lane) -> row = t + 8*half, col = l15.
  // Outputs are written once and never re-read: use non-temporal stores so the
  // 134 MB result stream does not perturb cache state for x / prototypes.
  const float temp = *tptr;
  float* __restrict__ out_sim  = out;
  float* __restrict__ out_dist = out + (size_t)N * M;
#pragma unroll
  for (int i = 0; i < 2; ++i)
#pragma unroll
    for (int j = 0; j < 2; ++j) {
      const int col = col_base + j * 16 + l15;
      const float pn = pnorm[col];
#pragma unroll
      for (int t = 0; t < 8; ++t) {
        const int row = row_base + i * 16 + half * 8 + t;
        const float dot = acc[i][j][t];
        float sq = xnorm[row] + pn - 2.0f * dot;
        sq = fmaxf(sq, 0.0f) + 1e-6f;
        const float dist = sqrtf(sq);
        const float sim  = expf(-temp * dist);
        const size_t o = (size_t)row * M + col;
        __builtin_nontemporal_store(sim,  out_sim  + o);
        __builtin_nontemporal_store(dist, out_dist + o);
      }
    }
}

// ---------------------------------------------------------------------------
// Launch: inputs are (x, prototypes, feature_weights, temperature).
// ---------------------------------------------------------------------------
extern "C" void kernel_launch(void* const* d_in, const int* in_sizes, int n_in,
                              void* d_out, int out_size, void* d_ws, size_t ws_size,
                              hipStream_t stream) {
  const float* x = (const float*)d_in[0];
  const float* p = (const float*)d_in[1];
  const float* w = (const float*)d_in[2];
  const float* t = (const float*)d_in[3];

  const int D = in_sizes[2];            // 256
  const int N = in_sizes[0] / D;        // 16384
  const int M = in_sizes[1] / D;        // 1024

  // Workspace layout (~18 MB): xhi, xlo [N*D f16]; phi, plo [M*D f16]; norms.
  _Float16* xhi = (_Float16*)d_ws;
  _Float16* xlo = xhi + (size_t)N * D;
  _Float16* phi = xlo + (size_t)N * D;
  _Float16* plo = phi + (size_t)M * D;
  float* xnorm  = (float*)(plo + (size_t)M * D);
  float* pnorm  = xnorm + N;

  prep_split_kernel<<<N, D, 0, stream>>>(x, w, xhi, xlo, xnorm, D);
  prep_split_kernel<<<M, D, 0, stream>>>(p, w, phi, plo, pnorm, D);

  dim3 grid(N / 128, M / 64);
  proto_wmma_kernel<<<grid, dim3(256), 0, stream>>>(
      xhi, xlo, phi, plo, xnorm, pnorm, t, (float*)d_out, N, M, D);
}
